// CoxSurvLoss_69999376990217
// MI455X (gfx1250) — compile-verified, matrix operand order checked by reading, not run on hardware
//
#include <hip/hip_runtime.h>
#include <hip/hip_bf16.h>

typedef __attribute__((ext_vector_type(16))) _Float16 v16h;
typedef __attribute__((ext_vector_type(8)))  float    v8f;
typedef __attribute__((ext_vector_type(4)))  int      v4i;
typedef __attribute__((ext_vector_type(2)))  int      v2i;

#define COX_EPS 1e-8f
#define CHUNK 1024
#define BLOCK_THREADS 256
#define WAVES_PER_BLOCK (BLOCK_THREADS / 32)

#define AS1 __attribute__((address_space(1)))
#define AS3 __attribute__((address_space(3)))

#if __has_builtin(__builtin_amdgcn_global_load_async_to_lds_b128) && \
    __has_builtin(__builtin_amdgcn_global_load_async_to_lds_b64)
#define HAVE_ASYNC_LDS 1
#else
#define HAVE_ASYNC_LDS 0
#endif

__device__ __forceinline__ void async_wait_all() {
#if __has_builtin(__builtin_amdgcn_s_wait_asynccnt)
    __builtin_amdgcn_s_wait_asynccnt(0);
#else
    asm volatile("s_wait_asynccnt 0" ::: "memory");
#endif
}

// ---------------------------------------------------------------------------
// Kernel 1: exp(risk) quantized to f16 (WMMA A-operand precision).
// ---------------------------------------------------------------------------
__global__ void cox_exp_kernel(const float* __restrict__ risk,
                               _Float16* __restrict__ exph, int n) {
    int i = blockIdx.x * blockDim.x + threadIdx.x;
    if (i < n) exph[i] = (_Float16)__expf(risk[i]);
}

// ---------------------------------------------------------------------------
// Kernel 2: denom[i] = sum_j (t[j] >= t[i]) * exp(risk[j]) via WMMA.
// One wave owns a 16-row tile; A = masked exp values, B = ones, so every
// column of the f32 accumulator carries the row sums. The j-stream is
// double-buffered in LDS via async global->LDS copies (ASYNCcnt) so the
// WMMA/VALU pipe overlaps the staging of the next chunk.
// ---------------------------------------------------------------------------
__global__ void cox_denom_kernel(const float* __restrict__ risk,
                                 const float* __restrict__ t,
                                 const float* __restrict__ ev,
                                 const _Float16* __restrict__ exph,
                                 float* __restrict__ partial,
                                 int n) {
    __shared__ __align__(16) float    s_t[2][CHUNK];
    __shared__ __align__(16) _Float16 s_e[2][CHUNK];

    const int lane     = threadIdx.x & 31;
    const int wave     = threadIdx.x >> 5;
    const int rowblock = blockIdx.x * WAVES_PER_BLOCK + wave;
    const int i0       = rowblock * 16;
    const int m        = lane & 15;   // row within tile owned by this lane (A layout)
    const int hf       = lane >> 4;   // half-wave selects K sub-range (A layout)
    const int x4       = threadIdx.x * 4;  // 4 elements/thread: 256*4 == CHUNK

    const float t_row = t[i0 + m];

    v16h ones;
#pragma unroll
    for (int e = 0; e < 16; ++e) ones[e] = (_Float16)1.0f;

    v8f acc = {};

    // Issue async copies of chunk starting at jc into LDS buffer `buf`.
    auto stage = [&](int jc, int buf) {
#if HAVE_ASYNC_LDS
        __builtin_amdgcn_global_load_async_to_lds_b128(
            (AS1 v4i*)(t + jc + x4), (AS3 v4i*)&s_t[buf][x4], 0, 0);
        __builtin_amdgcn_global_load_async_to_lds_b64(
            (AS1 v2i*)(exph + jc + x4), (AS3 v2i*)&s_e[buf][x4], 0, 0);
#else
        *(float4*)&s_t[buf][x4] = *(const float4*)(t + jc + x4);
        *(uint2*)&s_e[buf][x4]  = *(const uint2*)(exph + jc + x4);
#endif
    };

    auto compute_chunk = [&](const float* st, const _Float16* se, int len) {
        for (int j0 = 0; j0 < len; j0 += 32) {
            v16h a;
#pragma unroll
            for (int e = 0; e < 16; ++e) {
                // 16-bit A 16x32 layout: element e of lane L holds
                // K = e + (e & 8) + (L>>4)*8
                const int K = e + (e & 8) + hf * 8;
                const int j = j0 + K;
                const _Float16 v = se[j];
                a[e] = (st[j] >= t_row) ? v : (_Float16)0.0f;
            }
            acc = __builtin_amdgcn_wmma_f32_16x16x32_f16(
                false, a, false, ones, (short)0, acc, false, false);
        }
    };

    const int nfull = n / CHUNK;
    if (nfull > 0) {
        stage(0, 0);
        async_wait_all();
        __syncthreads();
        for (int c = 0; c < nfull; ++c) {
            const int buf = c & 1;
            if (c + 1 < nfull) stage((c + 1) * CHUNK, buf ^ 1);  // overlap
            compute_chunk(s_t[buf], s_e[buf], CHUNK);
            async_wait_all();   // own async copies for next chunk done
            __syncthreads();    // everyone done reading buf / writing buf^1
        }
    }
    // Tail (n not a multiple of CHUNK): synchronous staging, len multiple of 32.
    const int rem = n - nfull * CHUNK;
    if (rem > 0) {
        __syncthreads();
        for (int x = threadIdx.x; x < rem; x += BLOCK_THREADS) {
            s_t[0][x] = t[nfull * CHUNK + x];
            s_e[0][x] = exph[nfull * CHUNK + x];
        }
        __syncthreads();
        compute_chunk(s_t[0], s_e[0], rem);
    }

    // C/D layout: VGPR r holds row i0 + r + 8*hf, replicated across 16 lanes.
    float c = 0.0f;
#pragma unroll
    for (int r = 0; r < 8; ++r) {
        const int row = i0 + r + hf * 8;
        c += ev[row] * (risk[row] - __logf(acc[r] + COX_EPS));
    }
    // lanes 0-15 hold rows 0-7 contribution, lanes 16-31 rows 8-15: combine.
    const float other = __shfl_xor(c, 16, 32);
    if (lane == 0) partial[rowblock] = c + other;
}

// ---------------------------------------------------------------------------
// Kernel 3: deterministic fixed-order reduction of per-tile partials.
// ---------------------------------------------------------------------------
__global__ void cox_reduce_kernel(const float* __restrict__ partial,
                                  float* __restrict__ out, int nrb, int n) {
    __shared__ float s[256];
    const int tid = threadIdx.x;
    float v = 0.0f;
    for (int i = tid; i < nrb; i += 256) v += partial[i];
    s[tid] = v;
    __syncthreads();
    for (int stride = 128; stride > 0; stride >>= 1) {
        if (tid < stride) s[tid] += s[tid + stride];
        __syncthreads();
    }
    if (tid == 0) out[0] = -s[0] / (float)n;
}

extern "C" void kernel_launch(void* const* d_in, const int* in_sizes, int n_in,
                              void* d_out, int out_size, void* d_ws, size_t ws_size,
                              hipStream_t stream) {
    const float* risk = (const float*)d_in[0];
    const float* t    = (const float*)d_in[1];
    const float* ev   = (const float*)d_in[2];
    float* out        = (float*)d_out;
    const int n = in_sizes[0];            // 8192; assumed multiple of 128

    // workspace layout: [ exp_f16 : n*2 bytes | pad to 256 | partials : (n/16)*4 ]
    _Float16* exph = (_Float16*)d_ws;
    size_t off = ((size_t)n * sizeof(_Float16) + 255) & ~(size_t)255;
    float* partial = (float*)((char*)d_ws + off);
    const int nrb = n / 16;

    cox_exp_kernel<<<(n + 255) / 256, 256, 0, stream>>>(risk, exph, n);
    cox_denom_kernel<<<nrb / WAVES_PER_BLOCK, BLOCK_THREADS, 0, stream>>>(
        risk, t, ev, exph, partial, n);
    cox_reduce_kernel<<<1, 256, 0, stream>>>(partial, out, nrb, n);
}
